// Model_50646254354874
// MI455X (gfx1250) — compile-verified, matrix operand order checked by reading, not run on hardware
//
#include <hip/hip_runtime.h>
#include <hip/hip_bf16.h>

// ---------------------------------------------------------------------------
// CDNA5 (gfx1250) fused sliding-window GRU classifier.
// Matmuls: v_wmma_f32_16x16x32_bf16 (fp32 accumulate).
// Weights: pre-converted to bf16 once, staged into LDS with the gfx1250
// async global->LDS copy engine (ASYNCcnt / s_wait_asynccnt).
// ---------------------------------------------------------------------------

typedef __attribute__((ext_vector_type(16))) __bf16 v16bf;
typedef __attribute__((ext_vector_type(8)))  float  v8f;
typedef __attribute__((ext_vector_type(4)))  int    v4i;

#define DEV __device__ __forceinline__
#define ASL __attribute__((address_space(3)))

// ---- CDNA5 async global->LDS (16B per lane, tracked by ASYNCcnt) ----------
DEV void async_g2l_b128(const void* gp, void* lp) {
#if __has_builtin(__builtin_amdgcn_global_load_async_to_lds_b128)
    // param type per compiler diagnostic: v4i* (128-bit payload)
    __builtin_amdgcn_global_load_async_to_lds_b128((v4i*)gp, (v4i*)lp, 0, 0);
#else
    unsigned int loff = (unsigned int)(unsigned long long)(ASL char*)lp;
    asm volatile("global_load_async_to_lds_b128 %0, %1, off"
                 :: "v"(loff), "v"(gp) : "memory");
#endif
}

DEV void wait_async0() {
#if __has_builtin(__builtin_amdgcn_s_wait_asynccnt)
    __builtin_amdgcn_s_wait_asynccnt(0);
#else
    asm volatile("s_wait_asynccnt 0x0" ::: "memory");
#endif
}

// stage nbytes (multiple of 16B*blockDim) from global bf16 into LDS
DEV void stage_async(const __bf16* gsrc, __bf16* ldst, int nbytes,
                     int tid, int nthr) {
    for (int off = tid * 16; off < nbytes; off += nthr * 16)
        async_g2l_b128((const char*)gsrc + off, (char*)ldst + off);
}

// 16-bit A/B fragment element -> K index mapping (wave32, 16x16x32):
// lanes 0-15: elems 0..7 -> K=0..7, elems 8..15 -> K=16..23
// lanes16-31: elems 0..7 -> K=8..15, elems 8..15 -> K=24..31
DEV int fk(int hi, int e) { return (e & 7) | (hi << 3) | ((e >> 3) << 4); }

DEV v8f vzero8() {
    v8f v;
#pragma unroll
    for (int i = 0; i < 8; ++i) v[i] = 0.f;
    return v;
}

DEV v16bf bzero16() {
    v16bf v;
#pragma unroll
    for (int i = 0; i < 16; ++i) v[i] = (__bf16)0.f;
    return v;
}

DEV v8f wmma_bf16(v16bf a, v16bf b, v8f c) {
    // 8 args: (neg_a, A, neg_b, B, c_mod, C, reuse_a, reuse_b)
    return __builtin_amdgcn_wmma_f32_16x16x32_bf16(false, a, false, b,
                                                   (short)0, c, false, false);
}

// Fragment (A row = m, or B col = col) from bf16 LDS matrix [row][128];
// the fk() layout makes these 2x ds_load_b128 per fragment.
DEV v16bf load_frag_lds(const __bf16* base, int row, int c, int hi) {
    v16bf f;
#pragma unroll
    for (int e = 0; e < 16; ++e) f[e] = base[row * 128 + c * 32 + fk(hi, e)];
    return f;
}

DEV float sigm(float x) { return 1.f / (1.f + __expf(-x)); }

// ---------------------------------------------------------------------------
// Kernel P: one-shot fp32 -> bf16 weight conversion (amortized over all
// staging blocks; halves LDS-staging traffic).
// ---------------------------------------------------------------------------
__global__ void f32_to_bf16_kernel(const float* __restrict__ src,
                                   __bf16* __restrict__ dst, int n) {
    int i = blockIdx.x * blockDim.x + threadIdx.x;
    if (i < n) dst[i] = (__bf16)src[i];
}

// ---------------------------------------------------------------------------
// Kernel 1: deduped layer-0 input gates.
// gi0[b,t,:] = emb[x[b,t]] @ W_ih0^T + b_ih0   (computed once per token,
// not once per window-token: 16x less work). 4 waves/block, 16 tokens/wave.
// ---------------------------------------------------------------------------
__global__ void gates0_kernel(const int* __restrict__ xtok,
                              const float* __restrict__ emb,
                              const __bf16* __restrict__ wih0_bf,
                              const float* __restrict__ bih0,
                              float* __restrict__ gi0) {
    extern __shared__ char smem[];
    __bf16* wS = (__bf16*)smem;  // [384][128] bf16 weights (96KB)
    stage_async(wih0_bf, wS, 384 * 128 * 2, threadIdx.x, blockDim.x);
    wait_async0();
    __syncthreads();

    const int wave = threadIdx.x >> 5, lane = threadIdx.x & 31;
    const int hi = lane >> 4, mn = lane & 15;
    const int tt = blockIdx.x * 4 + wave;   // token tile (16 tokens)
    const int tok = tt * 16 + mn;           // A-fragment row token

    const float* erow = emb + (size_t)xtok[tok] * 128;
    v16bf a[4];
#pragma unroll
    for (int c = 0; c < 4; ++c) {
#pragma unroll
        for (int e = 0; e < 16; ++e)
            a[c][e] = (__bf16)erow[c * 32 + fk(hi, e)];
    }

    for (int nt = 0; nt < 24; ++nt) {  // 384 gate columns = 24 N-tiles
        v8f acc = vzero8();
#pragma unroll
        for (int c = 0; c < 4; ++c) {
            v16bf bb = load_frag_lds(wS, nt * 16 + mn, c, hi);
            acc = wmma_bf16(a[c], bb, acc);
        }
        const float bias = bih0[nt * 16 + mn];
#pragma unroll
        for (int r = 0; r < 8; ++r) {
            int rowtok = tt * 16 + r + 8 * hi;  // C-fragment row
            gi0[(size_t)rowtok * 384 + nt * 16 + mn] = acc[r] + bias;
        }
    }
}

// ---------------------------------------------------------------------------
// Kernel 2: fully fused 2-layer GRU + BatchNorm + MLP + scrambled max-pool.
// One block = 16 consecutive windows of one batch row; 4 waves, each wave
// owns 32 hidden columns (6 of 24 gate N-tiles). All state in LDS:
//   h0out [16 steps][16 win][128] bf16  64KB  (layer-0 trajectory)
//   h1    [16 win][128]           bf16   4KB  (layer-1 state)
//   wA/wI [384][128]              bf16 2x96KB (staged recurrent weights)
//   wM    [128][128]              bf16  32KB  (staged MLP weights)
// Total 292KB of the WGP's 320KB.
// ---------------------------------------------------------------------------
__global__ void fused_gru_kernel(const float* __restrict__ gi0,
                                 const __bf16* __restrict__ whh0_bf,
                                 const float* __restrict__ bhh0,
                                 const __bf16* __restrict__ wih1_bf,
                                 const float* __restrict__ bih1,
                                 const __bf16* __restrict__ whh1_bf,
                                 const float* __restrict__ bhh1,
                                 const float* __restrict__ bn_g,
                                 const float* __restrict__ bn_b,
                                 const float* __restrict__ bn_m,
                                 const float* __restrict__ bn_v,
                                 const __bf16* __restrict__ mlpw_bf,
                                 const float* __restrict__ mlp_b,
                                 float* __restrict__ pooled) {
    extern __shared__ char smem[];
    __bf16* h0out = (__bf16*)smem;                        // 65536 B
    __bf16* h1    = (__bf16*)(smem + 65536);              //  4096 B
    __bf16* wA    = (__bf16*)(smem + 69632);              // 98304 B
    __bf16* wI    = (__bf16*)(smem + 167936);             // 98304 B
    __bf16* wM    = (__bf16*)(smem + 266240);             // 32768 B

    const int b = blockIdx.x / 63, tile = blockIdx.x % 63;
    const int wave = threadIdx.x >> 5, lane = threadIdx.x & 31;
    const int hi = lane >> 4, mn = lane & 15;

    // Async-stage everything needed before/through layer 1 that does not
    // collide with layer-0's W_hh0 slot (wA): W_hh0, W_ih1, mlp_w.
    stage_async(whh0_bf, wA, 384 * 128 * 2, threadIdx.x, blockDim.x);
    stage_async(wih1_bf, wI, 384 * 128 * 2, threadIdx.x, blockDim.x);
    stage_async(mlpw_bf, wM, 128 * 128 * 2, threadIdx.x, blockDim.x);
    wait_async0();
    __syncthreads();

    // ---------------- layer 0: 16 recurrent steps ----------------
    for (int k = 0; k < 16; ++k) {
        v16bf a[4];
#pragma unroll
        for (int c = 0; c < 4; ++c)
            a[c] = (k == 0) ? bzero16()
                            : load_frag_lds(h0out + (k - 1) * 2048, mn, c, hi);

        // prefetch next step's gate rows while WMMAs run
        if (k < 15)
            __builtin_prefetch(
                gi0 + (size_t)(b * 1024 + tile * 16 + mn + k + 1) * 384, 0, 0);

        v8f acc[3][2];
#pragma unroll
        for (int g3 = 0; g3 < 3; ++g3)
#pragma unroll
            for (int j = 0; j < 2; ++j) acc[g3][j] = vzero8();

#pragma unroll
        for (int c = 0; c < 4; ++c)
#pragma unroll
            for (int g3 = 0; g3 < 3; ++g3)
#pragma unroll
                for (int j = 0; j < 2; ++j) {
                    v16bf bb = load_frag_lds(
                        wA, g3 * 128 + wave * 32 + j * 16 + mn, c, hi);
                    acc[g3][j] = wmma_bf16(a[c], bb, acc[g3][j]);
                }

#pragma unroll
        for (int j = 0; j < 2; ++j) {
            const int hcol = wave * 32 + j * 16 + mn;
            const float bhr = bhh0[hcol];
            const float bhz = bhh0[128 + hcol];
            const float bhn = bhh0[256 + hcol];
#pragma unroll
            for (int r = 0; r < 8; ++r) {
                const int m = r + 8 * hi;
                const int t = tile * 16 + m + k;  // token = window + step
                const float* gi = gi0 + (size_t)(b * 1024 + t) * 384;
                float rr = sigm(gi[hcol] + acc[0][j][r] + bhr);
                float zz = sigm(gi[128 + hcol] + acc[1][j][r] + bhz);
                float nn = tanhf(gi[256 + hcol] + rr * (acc[2][j][r] + bhn));
                float hp = (k == 0) ? 0.f
                         : (float)h0out[(k - 1) * 2048 + m * 128 + hcol];
                h0out[k * 2048 + m * 128 + hcol] =
                    (__bf16)((1.f - zz) * nn + zz * hp);
            }
        }
        __syncthreads();  // step k written before step k+1 reads it
    }

    // swap in layer-1 recurrent weights (wA dead after last barrier)
    stage_async(whh1_bf, wA, 384 * 128 * 2, threadIdx.x, blockDim.x);
    wait_async0();
    __syncthreads();

    // -------- layer 1 + BN + MLP + scrambled max-pool, per step --------
    for (int k = 0; k < 16; ++k) {
        v16bf ah[4], ax[4];
#pragma unroll
        for (int c = 0; c < 4; ++c) {
            ah[c] = (k == 0) ? bzero16() : load_frag_lds(h1, mn, c, hi);
            ax[c] = load_frag_lds(h0out + k * 2048, mn, c, hi);
        }
        __syncthreads();  // all h1 reads done before anyone rewrites it

        v8f gh[3][2], gx[3][2];
#pragma unroll
        for (int g3 = 0; g3 < 3; ++g3)
#pragma unroll
            for (int j = 0; j < 2; ++j) {
                gh[g3][j] = vzero8();
                gx[g3][j] = vzero8();
            }
#pragma unroll
        for (int c = 0; c < 4; ++c)
#pragma unroll
            for (int g3 = 0; g3 < 3; ++g3)
#pragma unroll
                for (int j = 0; j < 2; ++j) {
                    const int col = g3 * 128 + wave * 32 + j * 16 + mn;
                    gh[g3][j] = wmma_bf16(ah[c], load_frag_lds(wA, col, c, hi),
                                          gh[g3][j]);
                    gx[g3][j] = wmma_bf16(ax[c], load_frag_lds(wI, col, c, hi),
                                          gx[g3][j]);
                }

#pragma unroll
        for (int j = 0; j < 2; ++j) {
            const int hcol = wave * 32 + j * 16 + mn;
            const float bir = bih1[hcol], biz = bih1[128 + hcol],
                        bin = bih1[256 + hcol];
            const float bhr = bhh1[hcol], bhz = bhh1[128 + hcol],
                        bhn = bhh1[256 + hcol];
#pragma unroll
            for (int r = 0; r < 8; ++r) {
                const int m = r + 8 * hi;
                float rr = sigm(gx[0][j][r] + bir + gh[0][j][r] + bhr);
                float zz = sigm(gx[1][j][r] + biz + gh[1][j][r] + bhz);
                float nn = tanhf(gx[2][j][r] + bin + rr * (gh[2][j][r] + bhn));
                float hp = (k == 0) ? 0.f : (float)h1[m * 128 + hcol];
                h1[m * 128 + hcol] = (__bf16)((1.f - zz) * nn + zz * hp);
            }
        }
        __syncthreads();  // h1 fully written before MLP reads the full tile

        // BatchNorm over channel k folded into A-fragment construction
        const float scale = bn_g[k] * rsqrtf(bn_v[k] + 1e-5f);
        const float shift = bn_b[k] - bn_m[k] * scale;
        v16bf an[4];
#pragma unroll
        for (int c = 0; c < 4; ++c) {
#pragma unroll
            for (int e = 0; e < 16; ++e)
                an[c][e] = (__bf16)((float)h1[mn * 128 + c * 32 + fk(hi, e)] *
                                        scale + shift);
        }
        v8f mo[2];
        mo[0] = vzero8();
        mo[1] = vzero8();
#pragma unroll
        for (int c = 0; c < 4; ++c)
#pragma unroll
            for (int j = 0; j < 2; ++j)
                mo[j] = wmma_bf16(
                    an[c], load_frag_lds(wM, wave * 32 + j * 16 + mn, c, hi),
                    mo[j]);

        const float mb0 = mlp_b[wave * 32 + mn];
        const float mb1 = mlp_b[wave * 32 + 16 + mn];
#pragma unroll
        for (int r = 0; r < 8; ++r) {
            float v = fmaxf(fmaxf(mo[0][r] + mb0, 0.f),
                            fmaxf(mo[1][r] + mb1, 0.f));
            // rows r+8*hi share (s,k) across the 16 lanes of a half-wave:
            // reduce over columns, then one atomic per (row, step).
#pragma unroll
            for (int off = 1; off < 16; off <<= 1)
                v = fmaxf(v, __shfl_xor(v, off, 32));
            if (mn == 0) {
                const int m = r + 8 * hi;
                const int s = tile * 16 + m;
                const int g = (s * 16 + k) / 1008;  // reference reshape scramble
                atomicMax((unsigned int*)&pooled[b * 16 + g],
                          __float_as_uint(v));  // valid: values >= 0 post-ReLU
            }
        }
    }
}

// ---------------------------------------------------------------------------
// Kernel 0: zero the pooled max buffer (atomicMax identity for >=0 floats).
// ---------------------------------------------------------------------------
__global__ void zero_pooled_kernel(float* __restrict__ pooled) {
    if (threadIdx.x < 256) pooled[threadIdx.x] = 0.f;
}

// ---------------------------------------------------------------------------
// Kernel 3: tiny decoder  out = relu(pooled@dec1^T+b1)@dec2^T+b2  ([16,2]).
// ---------------------------------------------------------------------------
__global__ void decode_kernel(const float* __restrict__ pooled,
                              const float* __restrict__ d1w,
                              const float* __restrict__ d1b,
                              const float* __restrict__ d2w,
                              const float* __restrict__ d2b,
                              float* __restrict__ out) {
    const int tid = threadIdx.x;
    if (tid >= 32) return;
    const int bb = tid >> 1, c = tid & 1;
    float p[16];
#pragma unroll
    for (int kk = 0; kk < 16; ++kk) p[kk] = pooled[bb * 16 + kk];
    float acc = d2b[c];
    for (int h = 0; h < 128; ++h) {
        float t = d1b[h];
#pragma unroll
        for (int kk = 0; kk < 16; ++kk) t += p[kk] * d1w[h * 16 + kk];
        acc += d2w[c * 128 + h] * fmaxf(t, 0.f);
    }
    out[bb * 2 + c] = acc;
}

// ---------------------------------------------------------------------------
extern "C" void kernel_launch(void* const* d_in, const int* in_sizes, int n_in,
                              void* d_out, int out_size, void* d_ws,
                              size_t ws_size, hipStream_t stream) {
    (void)in_sizes; (void)n_in; (void)out_size; (void)ws_size;
    const int*   x    = (const int*)d_in[0];
    // d_in[1] = lens (unused by the reference)
    const float* emb  = (const float*)d_in[2];
    const float* wih  = (const float*)d_in[3];   // [2,384,128]
    const float* whh  = (const float*)d_in[4];   // [2,384,128]
    const float* bih  = (const float*)d_in[5];   // [2,384]
    const float* bhh  = (const float*)d_in[6];   // [2,384]
    const float* bng  = (const float*)d_in[7];
    const float* bnb  = (const float*)d_in[8];
    const float* bnm  = (const float*)d_in[9];
    const float* bnv  = (const float*)d_in[10];
    const float* mlpw = (const float*)d_in[11];  // [128,128]
    const float* mlpb = (const float*)d_in[12];
    const float* d1w  = (const float*)d_in[13];  // [128,16]
    const float* d1b  = (const float*)d_in[14];
    const float* d2w  = (const float*)d_in[15];  // [2,128]
    const float* d2b  = (const float*)d_in[16];

    // workspace layout
    char* ws = (char*)d_ws;
    float*  gi0     = (float*)ws;                              // 25,165,824 B
    float*  pooled  = (float*)(ws + 25165824);                 //      1,024 B
    __bf16* wih0_bf = (__bf16*)(ws + 25166848);                //     98,304 B
    __bf16* whh0_bf = (__bf16*)(ws + 25166848 + 98304);        //     98,304 B
    __bf16* wih1_bf = (__bf16*)(ws + 25166848 + 2 * 98304);    //     98,304 B
    __bf16* whh1_bf = (__bf16*)(ws + 25166848 + 3 * 98304);    //     98,304 B
    __bf16* mlpw_bf = (__bf16*)(ws + 25166848 + 4 * 98304);    //     32,768 B

    zero_pooled_kernel<<<1, 256, 0, stream>>>(pooled);

    // one-shot bf16 weight conversion (amortized across all staging blocks)
    f32_to_bf16_kernel<<<192, 256, 0, stream>>>(wih, wih0_bf, 49152);
    f32_to_bf16_kernel<<<192, 256, 0, stream>>>(whh, whh0_bf, 49152);
    f32_to_bf16_kernel<<<192, 256, 0, stream>>>(wih + 49152, wih1_bf, 49152);
    f32_to_bf16_kernel<<<192, 256, 0, stream>>>(whh + 49152, whh1_bf, 49152);
    f32_to_bf16_kernel<<<64, 256, 0, stream>>>(mlpw, mlpw_bf, 16384);

    // layer-0 input gates, deduped per token: 1024 token-tiles, 4 waves/block
    gates0_kernel<<<256, 128, 384 * 128 * 2, stream>>>(x, emb, wih0_bf, bih,
                                                       gi0);

    // fused 2-layer GRU + BN + MLP + pool: 16 batches x 63 window-tiles
    fused_gru_kernel<<<16 * 63, 128, 299008, stream>>>(
        gi0, whh0_bf, bhh, wih1_bf, bih + 384, whh1_bf, bhh + 384,
        bng, bnb, bnm, bnv, mlpw_bf, mlpb, pooled);

    decode_kernel<<<1, 32, 0, stream>>>(pooled, d1w, d1b, d2w, d2b,
                                        (float*)d_out);
}
